// SVDHead_20675972563723
// MI455X (gfx1250) — matrix operand order repaired and checked by reading, not compiled, not run on hardware
//
#include <hip/hip_runtime.h>
#include <hip/hip_bf16.h>
#include <math.h>

typedef __attribute__((ext_vector_type(16))) __bf16 v16bf;
typedef __attribute__((ext_vector_type(8)))  float  v8f;

#define NPTS 2048
#define DEMB 512
#define BATCH 4
#define LDA 520           // padded row stride (ushorts) -> 1040B, 16B aligned
#define ROWS 64           // src rows per workgroup
#define BLK 128           // 4 waves of 32
#define NTILES (NPTS / 16)
#define SCALE_IN  0.04419417382415922f   // 1/sqrt(512)
#define SCALE_POST 10000.0f
#define EPSF 1e-6f

union FragU { uint4 q[2]; unsigned int u[8]; v16bf v; };

static __device__ __forceinline__ unsigned short f2bf(float x) {
  unsigned int u = __float_as_uint(x);
  unsigned int r = u + 0x7FFFu + ((u >> 16) & 1u);   // round-to-nearest-even
  return (unsigned short)(r >> 16);
}

// ---------------------------------------------------------------------------
// Kernel 1: fused scores-GEMM (bf16 WMMA) + softmax(10000x) + argmax + corr_tgt
// Per-lane online softmax (no cross-lane traffic in the hot loop); lane states
// merged once at the end. Double-buffered B staging: one barrier per tile.
// ---------------------------------------------------------------------------
__global__ void attn_kernel(const float* __restrict__ srcE, const float* __restrict__ tgtE,
                            const float* __restrict__ tgtP, float* __restrict__ corrW,
                            float* __restrict__ outF) {
  extern __shared__ char smem[];
  unsigned short* sA = (unsigned short*)smem;        // 64 x LDA bf16
  unsigned short* sB = sA + ROWS * LDA;              // 2 x 16 x LDA bf16 (double buffer)
  const int b    = blockIdx.y;
  const int n0   = blockIdx.x * ROWS;
  const int tid  = threadIdx.x;
  const int lane = tid & 31, wave = tid >> 5;
  const int hf   = lane >> 4, l16 = lane & 15;

  // ---- stage A tile (64 rows x 512 K), scale by 1/sqrt(d), cvt to bf16 ----
  for (int i = 0; i < (ROWS * DEMB) / (BLK * 4); ++i) {
    int idx = tid + i * BLK;
    int d = idx >> 4, r4 = (idx & 15) * 4;
    float4 v = *(const float4*)&srcE[((size_t)b * DEMB + d) * NPTS + n0 + r4];
    sA[(r4 + 0) * LDA + d] = f2bf(v.x * SCALE_IN);
    sA[(r4 + 1) * LDA + d] = f2bf(v.y * SCALE_IN);
    sA[(r4 + 2) * LDA + d] = f2bf(v.z * SCALE_IN);
    sA[(r4 + 3) * LDA + d] = f2bf(v.w * SCALE_IN);
  }

  // ---- prologue: stage B tile 0 into buffer 0 ----
  for (int i = 0; i < (16 * DEMB) / (BLK * 4); ++i) {
    int idx = tid + i * BLK;
    int d = idx >> 2, c4 = (idx & 3) * 4;
    float4 v = *(const float4*)&tgtE[((size_t)b * DEMB + d) * NPTS + c4];
    sB[(c4 + 0) * LDA + d] = f2bf(v.x);
    sB[(c4 + 1) * LDA + d] = f2bf(v.y);
    sB[(c4 + 2) * LDA + d] = f2bf(v.z);
    sB[(c4 + 3) * LDA + d] = f2bf(v.w);
  }

  const int rowbase = wave * 16;
  float run_max[8], lsum[8], ax[8], ay[8], az[8], best[8];
  int   bidx[8];
#pragma unroll
  for (int r = 0; r < 8; ++r) {
    run_max[r] = -1e30f; lsum[r] = 0.f; ax[r] = ay[r] = az[r] = 0.f;
    best[r] = -1e30f; bidx[r] = 0;
  }

  const unsigned short* arow = sA + (rowbase + l16) * LDA;  // A: M = l16 (both halves)

  for (int mt = 0; mt < NTILES; ++mt) {
    __syncthreads();   // buf[mt&1] staged; everyone done with buf[(mt+1)&1]

    // ---- stage NEXT B tile into the alternate buffer (overlaps compute) ----
    if (mt + 1 < NTILES) {
      unsigned short* bn = sB + ((mt + 1) & 1) * 16 * LDA;
      for (int i = 0; i < (16 * DEMB) / (BLK * 4); ++i) {
        int idx = tid + i * BLK;
        int d = idx >> 2, c4 = (idx & 3) * 4;
        const float* gp = &tgtE[((size_t)b * DEMB + d) * NPTS + (mt + 1) * 16 + c4];
        float4 v = *(const float4*)gp;
        if (mt + 2 < NTILES) __builtin_prefetch(gp + 16, 0, 1);  // tile after next
        bn[(c4 + 0) * LDA + d] = f2bf(v.x);
        bn[(c4 + 1) * LDA + d] = f2bf(v.y);
        bn[(c4 + 2) * LDA + d] = f2bf(v.z);
        bn[(c4 + 3) * LDA + d] = f2bf(v.w);
      }
    }

    // ---- 16x16 score tile via WMMA over K = 512 ----
    const unsigned short* brow = sB + (mt & 1) * 16 * LDA + l16 * LDA;
    v8f c = {0.f, 0.f, 0.f, 0.f, 0.f, 0.f, 0.f, 0.f};
#pragma unroll 4
    for (int ks = 0; ks < DEMB / 32; ++ks) {
      int kc = ks * 32;
      FragU fa, fb;
      fa.q[0] = *(const uint4*)(arow + kc + 8 * hf);        // K = 8h .. 8h+7
      fa.q[1] = *(const uint4*)(arow + kc + 8 * hf + 16);   // K = 16+8h ..
      fb.q[0] = *(const uint4*)(brow + kc + 8 * hf);        // K = 8h .. 8h+7
      fb.q[1] = *(const uint4*)(brow + kc + 8 * hf + 16);   // K = 16+8h ..
      c = __builtin_amdgcn_wmma_f32_16x16x32_bf16(false, fa.v, false, fb.v,
                                                  (short)0, c, false, false);
    }

    // ---- per-lane online softmax / argmax / weighted tgt accumulation ----
    int m = mt * 16 + l16;            // this lane's score column
    float tx = tgtP[((size_t)b * 3 + 0) * NPTS + m];
    float ty = tgtP[((size_t)b * 3 + 1) * NPTS + m];
    float tz = tgtP[((size_t)b * 3 + 2) * NPTS + m];
#pragma unroll
    for (int r = 0; r < 8; ++r) {
      float s = c[r] * SCALE_POST;
      if (s > best[r]) { best[r] = s; bidx[r] = m; }
      float om   = run_max[r];
      float nm   = fmaxf(om, s);
      float corr = __expf(om - nm);        // == 1 when max unchanged
      float p    = __expf(s - nm);
      run_max[r] = nm;
      lsum[r] = lsum[r] * corr + p;
      ax[r]   = ax[r]   * corr + p * tx;
      ay[r]   = ay[r]   * corr + p * ty;
      az[r]   = az[r]   * corr + p * tz;
    }
  }

  // ---- merge the 16 per-lane softmax states (butterfly, once) ----
#pragma unroll
  for (int r = 0; r < 8; ++r) {
    float mm = run_max[r], l = lsum[r], x = ax[r], y = ay[r], z = az[r];
    float bv = best[r]; int bi = bidx[r];
#pragma unroll
    for (int msk = 1; msk < 16; msk <<= 1) {
      float om = __shfl_xor(mm, msk, 32);
      float ol = __shfl_xor(l,  msk, 32);
      float ox = __shfl_xor(x,  msk, 32);
      float oy = __shfl_xor(y,  msk, 32);
      float oz = __shfl_xor(z,  msk, 32);
      float ov = __shfl_xor(bv, msk, 32);
      int   oi = __shfl_xor(bi, msk, 32);
      float nm = fmaxf(mm, om);
      float c1 = __expf(mm - nm), c2 = __expf(om - nm);
      l = l * c1 + ol * c2;
      x = x * c1 + ox * c2;
      y = y * c1 + oy * c2;
      z = z * c1 + oz * c2;
      mm = nm;
      if (ov > bv || (ov == bv && oi < bi)) { bv = ov; bi = oi; }
    }
    if (l16 == 0) {
      int n = n0 + rowbase + r + 8 * hf;     // lanes 0/16 own rows r / r+8
      float inv = 1.0f / l;
      corrW[((size_t)b * NPTS + n) * 3 + 0] = x * inv;
      corrW[((size_t)b * NPTS + n) * 3 + 1] = y * inv;
      corrW[((size_t)b * NPTS + n) * 3 + 2] = z * inv;
      outF[48 + (size_t)b * NPTS + n] = (float)bi;   // corres
    }
  }
}

// ---------------------------------------------------------------------------
// Kernel 2: _gfm per-point loss (knn on src pts, 45 triangle pairs)
// ---------------------------------------------------------------------------
__global__ void gfm_kernel(const float* __restrict__ srcP, const float* __restrict__ corrW,
                           float* __restrict__ lossW) {
  __shared__ float sp[NPTS * 3];
  __shared__ float ct[NPTS * 3];
  int b = blockIdx.x >> 3;
  int n = (blockIdx.x & 7) * 256 + threadIdx.x;
  for (int i = threadIdx.x; i < NPTS; i += 256) {
    sp[i * 3 + 0] = srcP[((size_t)b * 3 + 0) * NPTS + i];
    sp[i * 3 + 1] = srcP[((size_t)b * 3 + 1) * NPTS + i];
    sp[i * 3 + 2] = srcP[((size_t)b * 3 + 2) * NPTS + i];
    ct[i * 3 + 0] = corrW[((size_t)b * NPTS + i) * 3 + 0];
    ct[i * 3 + 1] = corrW[((size_t)b * NPTS + i) * 3 + 1];
    ct[i * 3 + 2] = corrW[((size_t)b * NPTS + i) * 3 + 2];
  }
  __syncthreads();

  float px = sp[n * 3 + 0], py = sp[n * 3 + 1], pz = sp[n * 3 + 2];
  float qx = ct[n * 3 + 0], qy = ct[n * 3 + 1], qz = ct[n * 3 + 2];

  // top-10 smallest d2 (== top-10 largest gaussian weight), stable on ties
  float nd[10]; int ni[10];
#pragma unroll
  for (int k = 0; k < 10; ++k) { nd[k] = 1e30f; ni[k] = 0; }
  for (int m = 0; m < NPTS; ++m) {
    float dx = px - sp[m * 3 + 0], dy = py - sp[m * 3 + 1], dz = pz - sp[m * 3 + 2];
    float d2 = dx * dx + dy * dy + dz * dz;
    if (d2 < nd[9]) {
      float cd = d2; int ci = m;
#pragma unroll
      for (int k = 0; k < 10; ++k)
        if (cd < nd[k]) { float td = nd[k]; int ti = ni[k];
                          nd[k] = cd; ni[k] = ci; cd = td; ci = ti; }
    }
  }

  float loss45[45], reg45[45];
  int k = 0;
  for (int a = 0; a < 9; ++a)
    for (int b2 = a + 1; b2 < 10; ++b2, ++k) {
      int i = ni[a], j = ni[b2];
      // src triangle area
      float e1x = sp[i*3+0]-px, e1y = sp[i*3+1]-py, e1z = sp[i*3+2]-pz;
      float e2x = sp[j*3+0]-px, e2y = sp[j*3+1]-py, e2z = sp[j*3+2]-pz;
      float cx = e1y*e2z - e1z*e2y, cy = e1z*e2x - e1x*e2z, cz = e1x*e2y - e1y*e2x;
      float a_s = 0.5f * sqrtf(cx*cx + cy*cy + cz*cz);
      // corr_tgt triangle area
      float f1x = ct[i*3+0]-qx, f1y = ct[i*3+1]-qy, f1z = ct[i*3+2]-qz;
      float f2x = ct[j*3+0]-qx, f2y = ct[j*3+1]-qy, f2z = ct[j*3+2]-qz;
      float gx = f1y*f2z - f1z*f2y, gy = f1z*f2x - f1x*f2z, gz = f1x*f2y - f1y*f2x;
      float a_t = 0.5f * sqrtf(gx*gx + gy*gy + gz*gz);

      float diff = a_s - a_t - EPSF;
      float num  = diff * diff + 2.0f * EPSF * EPSF;
      float den  = a_s + a_t + EPSF;  den = den * den + 2.0f * EPSF * EPSF;
      loss45[k] = num / den;
      reg45[k]  = sqrtf(num);
    }

  // sort loss ascending (reference sorts BEFORE adding unsorted reg)
  for (int i2 = 1; i2 < 45; ++i2) {
    float key = loss45[i2]; int j2 = i2 - 1;
    while (j2 >= 0 && loss45[j2] > key) { loss45[j2 + 1] = loss45[j2]; --j2; }
    loss45[j2 + 1] = key;
  }
  for (int k2 = 0; k2 < 45; ++k2) loss45[k2] += 0.1f * reg45[k2];

  float tmp[45];
  for (int k2 = 0; k2 < 45; ++k2) tmp[k2] = loss45[k2];
  for (int i2 = 1; i2 < 45; ++i2) {
    float key = tmp[i2]; int j2 = i2 - 1;
    while (j2 >= 0 && tmp[j2] > key) { tmp[j2 + 1] = tmp[j2]; --j2; }
    tmp[j2 + 1] = key;
  }
  float med = tmp[22];
  for (int k2 = 0; k2 < 45; ++k2) if (loss45[k2] > 3.0f * med) loss45[k2] = 0.0f;

  float acc = 0.0f;
  for (int k2 = 0; k2 < 10; ++k2) acc += sqrtf(loss45[k2] + EPSF);
  lossW[(size_t)b * NPTS + n] = acc * 0.1f;
}

// ---------------------------------------------------------------------------
// Kernel 3/4: per-batch min, then binary weight
// ---------------------------------------------------------------------------
__global__ void minred_kernel(const float* __restrict__ lossW, float* __restrict__ lmin) {
  __shared__ float red[256];
  int b = blockIdx.x, t = threadIdx.x;
  float v = 1e30f;
  for (int i = t; i < NPTS; i += 256) v = fminf(v, lossW[(size_t)b * NPTS + i]);
  red[t] = v; __syncthreads();
  for (int s = 128; s > 0; s >>= 1) { if (t < s) red[t] = fminf(red[t], red[t + s]); __syncthreads(); }
  if (t == 0) lmin[b] = red[0];
}

__global__ void weight_kernel(const float* __restrict__ lossW, const float* __restrict__ lmin,
                              float* __restrict__ wW, float* __restrict__ outF) {
  int b = blockIdx.x >> 3;
  int n = (blockIdx.x & 7) * 256 + threadIdx.x;
  float l  = lossW[(size_t)b * NPTS + n] - lmin[b];
  float w2 = 2.0f / (1.0f + __expf(20.0f * l));       // 2*sigmoid(-20*l)
  float w  = (w2 > 0.5f) ? 1.0f : 0.0f;
  wW[(size_t)b * NPTS + n] = w;
  outF[48 + BATCH * NPTS /*=8240 base*/ + (size_t)b * NPTS + n] = w;
}

// ---------------------------------------------------------------------------
// Kernel 5: weighted Procrustes (block reductions + thread-0 3x3 SVD)
// ---------------------------------------------------------------------------
__global__ void procrustes_kernel(const float* __restrict__ srcP, const float* __restrict__ tgtP,
                                  const float* __restrict__ wW, const float* __restrict__ corres,
                                  float* __restrict__ outF) {
  __shared__ float red[256];
  int b = blockIdx.x, t = threadIdx.x;

  auto bsum = [&](float v) -> float {
    red[t] = v; __syncthreads();
    for (int s = 128; s > 0; s >>= 1) { if (t < s) red[t] += red[t + s]; __syncthreads(); }
    float r = red[0]; __syncthreads(); return r;
  };

  float pw = 0.f, pxs[3] = {0.f, 0.f, 0.f}, pys[3] = {0.f, 0.f, 0.f};
  for (int n = t; n < NPTS; n += 256) {
    float w = wW[(size_t)b * NPTS + n];
    int  ci = (int)corres[(size_t)b * NPTS + n];
    for (int c = 0; c < 3; ++c) {
      pxs[c] += w * srcP[((size_t)b * 3 + c) * NPTS + n];
      pys[c] += w * tgtP[((size_t)b * 3 + c) * NPTS + ci];
    }
    pw += w;
  }
  float W1  = bsum(pw);
  float inv = 1.0f / (W1 + 1e-7f);
  float mux[3], muy[3];
  for (int c = 0; c < 3; ++c) { mux[c] = bsum(pxs[c]) * inv; muy[c] = bsum(pys[c]) * inv; }

  float S[9] = {0.f,0.f,0.f,0.f,0.f,0.f,0.f,0.f,0.f};
  for (int n = t; n < NPTS; n += 256) {
    float w = wW[(size_t)b * NPTS + n] * inv;
    int  ci = (int)corres[(size_t)b * NPTS + n];
    float X[3], Y[3];
    for (int c = 0; c < 3; ++c) {
      X[c] = srcP[((size_t)b * 3 + c) * NPTS + n]  - mux[c];
      Y[c] = tgtP[((size_t)b * 3 + c) * NPTS + ci] - muy[c];
    }
    for (int i = 0; i < 3; ++i)
      for (int j = 0; j < 3; ++j) S[i * 3 + j] += w * Y[i] * X[j];
  }
  for (int k = 0; k < 9; ++k) S[k] = bsum(S[k]);

  if (t == 0) {
    float A_[3][3], Bm[3][3], V[3][3] = {{1,0,0},{0,1,0},{0,0,1}};
    for (int i = 0; i < 3; ++i) for (int j = 0; j < 3; ++j) A_[i][j] = S[i * 3 + j];
    for (int i = 0; i < 3; ++i) for (int j = 0; j < 3; ++j) {
      float s2 = 0.f; for (int k = 0; k < 3; ++k) s2 += A_[k][i] * A_[k][j];
      Bm[i][j] = s2;
    }
    const int PQ[3][2] = {{0,1},{0,2},{1,2}};
    for (int sw = 0; sw < 12; ++sw)
      for (int r = 0; r < 3; ++r) {
        int p = PQ[r][0], q = PQ[r][1];
        float apq = Bm[p][q];
        if (fabsf(apq) > 1e-20f) {
          float theta = (Bm[q][q] - Bm[p][p]) / (2.0f * apq);
          float tt = copysignf(1.0f, theta) / (fabsf(theta) + sqrtf(1.0f + theta * theta));
          float cc = rsqrtf(1.0f + tt * tt), ss = tt * cc;
          for (int k = 0; k < 3; ++k) {
            float bkp = Bm[k][p], bkq = Bm[k][q];
            Bm[k][p] = cc * bkp - ss * bkq; Bm[k][q] = ss * bkp + cc * bkq;
          }
          for (int k = 0; k < 3; ++k) {
            float bpk = Bm[p][k], bqk = Bm[q][k];
            Bm[p][k] = cc * bpk - ss * bqk; Bm[q][k] = ss * bpk + cc * bqk;
          }
          for (int k = 0; k < 3; ++k) {
            float vkp = V[k][p], vkq = V[k][q];
            V[k][p] = cc * vkp - ss * vkq; V[k][q] = ss * vkp + cc * vkq;
          }
        }
      }
    float lam[3] = {Bm[0][0], Bm[1][1], Bm[2][2]};
    for (int i = 0; i < 2; ++i) {
      int mx = i;
      for (int j = i + 1; j < 3; ++j) if (lam[j] > lam[mx]) mx = j;
      if (mx != i) {
        float tl = lam[i]; lam[i] = lam[mx]; lam[mx] = tl;
        for (int k = 0; k < 3; ++k) { float tv = V[k][i]; V[k][i] = V[k][mx]; V[k][mx] = tv; }
      }
    }
    float U[3][3];
    for (int i = 0; i < 3; ++i) {
      float u0 = 0.f, u1 = 0.f, u2 = 0.f;
      for (int k = 0; k < 3; ++k) {
        u0 += A_[0][k] * V[k][i]; u1 += A_[1][k] * V[k][i]; u2 += A_[2][k] * V[k][i];
      }
      float nrm = sqrtf(u0 * u0 + u1 * u1 + u2 * u2) + 1e-20f;
      U[0][i] = u0 / nrm; U[1][i] = u1 / nrm; U[2][i] = u2 / nrm;
    }
    float detU = U[0][0]*(U[1][1]*U[2][2]-U[1][2]*U[2][1])
               - U[0][1]*(U[1][0]*U[2][2]-U[1][2]*U[2][0])
               + U[0][2]*(U[1][0]*U[2][1]-U[1][1]*U[2][0]);
    float detV = V[0][0]*(V[1][1]*V[2][2]-V[1][2]*V[2][1])
               - V[0][1]*(V[1][0]*V[2][2]-V[1][2]*V[2][0])
               + V[0][2]*(V[1][0]*V[2][1]-V[1][1]*V[2][0]);
    float sgn = (detU * detV < 0.f) ? -1.f : 1.f;
    float d3[3] = {1.f, 1.f, sgn};
    float R[3][3];
    for (int i = 0; i < 3; ++i)
      for (int j = 0; j < 3; ++j) {
        float s2 = 0.f;
        for (int k = 0; k < 3; ++k) s2 += U[i][k] * d3[k] * V[j][k];
        R[i][j] = s2;
        outF[(size_t)b * 9 + i * 3 + j] = s2;
      }
    for (int i = 0; i < 3; ++i) {
      float ti = muy[i];
      for (int j = 0; j < 3; ++j) ti -= R[i][j] * mux[j];
      outF[36 + (size_t)b * 3 + i] = ti;
    }
  }
}

// ---------------------------------------------------------------------------
extern "C" void kernel_launch(void* const* d_in, const int* in_sizes, int n_in,
                              void* d_out, int out_size, void* d_ws, size_t ws_size,
                              hipStream_t stream) {
  const float* srcE = (const float*)d_in[0];
  const float* tgtE = (const float*)d_in[1];
  const float* srcP = (const float*)d_in[2];
  const float* tgtP = (const float*)d_in[3];
  float* outF = (float*)d_out;
  float* ws   = (float*)d_ws;

  float* corrW = ws;                          // B*N*3 = 24576
  float* lossW = corrW + BATCH * NPTS * 3;    // B*N
  float* wW    = lossW + BATCH * NPTS;        // B*N
  float* lmin  = wW    + BATCH * NPTS;        // B

  dim3 gA(NPTS / ROWS, BATCH);                // 32 x 4
  size_t smemA = (size_t)(ROWS + 32) * LDA * sizeof(unsigned short);  // ~100 KB (A + 2x B)
  attn_kernel<<<gA, dim3(BLK), smemA, stream>>>(srcE, tgtE, tgtP, corrW, outF);
  gfm_kernel<<<dim3(BATCH * 8), dim3(256), 0, stream>>>(srcP, corrW, lossW);
  minred_kernel<<<dim3(BATCH), dim3(256), 0, stream>>>(lossW, lmin);
  weight_kernel<<<dim3(BATCH * 8), dim3(256), 0, stream>>>(lossW, lmin, wW, outF);
  procrustes_kernel<<<dim3(BATCH), dim3(256), 0, stream>>>(srcP, tgtP, wW, outF + 48, outF);
}
